// ST_LLM_88055419502627
// MI455X (gfx1250) — matrix operand order repaired and checked by reading, not compiled, not run on hardware
//
#include <hip/hip_runtime.h>

// ---------------- static model config (matches reference) ----------------
#define Bh    8
#define TIN   12
#define NN    250
#define DIN   3
#define DLc   36           // D_IN * T_IN
#define TCNC  64
#define DEEP  1024
#define HID   2048
#define NH    16
#define HD    128
#define DFF   10944
#define MFF   1408
#define SFF   2816
#define NEXP  8
#define TIMEN 48
#define OUTL  12
#define NTOK  (Bh*NN)      // 2000 tokens

// ---------------- CDNA5 WMMA types ----------------
typedef __attribute__((ext_vector_type(16))) __bf16 v16bf;
typedef __attribute__((ext_vector_type(8)))  float  v8f;

__device__ __forceinline__ unsigned short f2bf(float f) {
  unsigned u = __float_as_uint(f);
  u += 0x7FFFu + ((u >> 16) & 1u);        // round-to-nearest-even
  return (unsigned short)(u >> 16);
}
__device__ __forceinline__ float bf2f(unsigned short s) {
  return __uint_as_float(((unsigned)s) << 16);
}

// =========================================================================
// Core GEMM: C[M,N] = beta*C + scale[m] * (A_bf16[M,K] x W_f32[N,K]^T + bias)
//
// CDNA5 data path:
//  * A tile (128x32 bf16, already bf16 in workspace) is DMA'd into a
//    double-buffered LDS tile with GLOBAL_LOAD_ASYNC_TO_LDS_B128
//    (ASYNCcnt-tracked; no VGPR round trip), overlapped with compute and
//    synchronized via s_wait_asynccnt + workgroup barrier.
//  * W tile (64x32 fp32) is cooperatively loaded once per WG, converted to
//    bf16 once (not per-wave), ds_store'd, and read back as fragments with
//    ds_load_b128.
//  * 8 waves (wave32): 4(M) x 2(N); wave tile 32x32 = 2x2
//    v_wmma_f32_16x16x32_bf16, fp32 accumulation.
// =========================================================================
union FragA { v16bf v; uint4 q[2]; };
union FragB { v16bf v; uint4 q[2]; unsigned short s[16]; };
union FragC { v8f   v; float f[8]; };

__global__ __launch_bounds__(256) void k_gemm(
    const unsigned short* __restrict__ A, const float* __restrict__ W,
    float* __restrict__ C, const float* __restrict__ bias,
    const float* __restrict__ scale, int M, int N, int K, int beta)
{
  __shared__ unsigned short smA[2][128][32];   // 2 x 8 KB, async-DMA target
  __shared__ unsigned short smB[64][32];       // 4 KB, converted weights

  const int tid  = threadIdx.x;
  const int lane = tid & 31;
  const int wave = tid >> 5;                   // 0..7
  const int wm   = wave >> 1, wn = wave & 1;   // 4 (M) x 2 (N)
  const int half = lane >> 4;
  const int l16  = lane & 15;
  const int rowBase = blockIdx.y * 128;
  const int colBase = blockIdx.x * 64;

  // ---- A async staging map: 128 rows x 64 B = 512 16-B chunks; thread owns
  //      chunks {tid, tid+256}: row = chunk/4, K sub-offset = (chunk%4)*8.
  int ar0 = rowBase + (tid >> 2);        if (ar0 >= M) ar0 = M - 1;
  int ar1 = rowBase + ((tid + 256) >> 2); if (ar1 >= M) ar1 = M - 1;
  const int ac = (tid & 3) * 8;
  const unsigned ldsA0 = (unsigned)(size_t)&smA[0][tid >> 2][ac];
  const unsigned ldsA1 = (unsigned)(size_t)&smA[0][(tid + 256) >> 2][ac];
  const unsigned bufStride = (unsigned)sizeof(smA[0]);   // 8192 B

  // ---- B staging map: 64 rows x 32 fp32 = 512 4-float chunks; thread owns
  //      chunks {tid, tid+256}: rows tid/8 and 32+tid/8, pos (tid%8)*4.
  int wr0 = colBase + (tid >> 3);        if (wr0 >= N) wr0 = N - 1;
  int wr1 = colBase + 32 + (tid >> 3);   if (wr1 >= N) wr1 = N - 1;
  const int bpp = (tid & 7) * 4;

  FragC acc[2][2];
  #pragma unroll
  for (int i = 0; i < 2; i++)
    #pragma unroll
    for (int j = 0; j < 2; j++)
      #pragma unroll
      for (int r = 0; r < 8; r++) acc[i][j].f[r] = 0.f;

  auto issueA = [&](int kb, int buf) {
    const unsigned short* g0 = A + (size_t)ar0 * K + kb + ac;
    const unsigned short* g1 = A + (size_t)ar1 * K + kb + ac;
    unsigned d0 = ldsA0 + (unsigned)buf * bufStride;
    unsigned d1 = ldsA1 + (unsigned)buf * bufStride;
    asm volatile("global_load_async_to_lds_b128 %0, %1, off"
                 :: "v"(d0), "v"(g0) : "memory");
    asm volatile("global_load_async_to_lds_b128 %0, %1, off"
                 :: "v"(d1), "v"(g1) : "memory");
  };

  const int KT = K >> 5;                 // K is a multiple of 32 for all users
  issueA(0, 0);

  for (int kt = 0; kt < KT; kt++) {
    const int kb = kt << 5;
    if (kt + 1 < KT) issueA(kb + 32, (kt + 1) & 1);

    // stage + convert W tile (once per WG instead of once per wave)
    {
      const float* p0 = W + (size_t)wr0 * K + kb + bpp;
      const float* p1 = W + (size_t)wr1 * K + kb + bpp;
      float4 f0 = *(const float4*)p0;
      float4 f1 = *(const float4*)p1;
      unsigned short* s0 = &smB[tid >> 3][bpp];
      unsigned short* s1 = &smB[32 + (tid >> 3)][bpp];
      s0[0] = f2bf(f0.x); s0[1] = f2bf(f0.y); s0[2] = f2bf(f0.z); s0[3] = f2bf(f0.w);
      s1[0] = f2bf(f1.x); s1[1] = f2bf(f1.y); s1[2] = f2bf(f1.z); s1[3] = f2bf(f1.w);
      if (kb + 32 < K) {                 // global_prefetch_b8 next weight slab
        __builtin_prefetch(p0 + 32, 0, 1);
        __builtin_prefetch(p1 + 32, 0, 1);
      }
    }

    // wait for this K-step's A DMA (2 async instrs per tile per wave,
    // in-order completion; one extra tile may remain in flight)
    if (kt + 1 < KT) asm volatile("s_wait_asynccnt 0x2" ::: "memory");
    else             asm volatile("s_wait_asynccnt 0x0" ::: "memory");
    __syncthreads();

    // fragment loads from LDS (ds_load_b128)
    FragA a[2]; FragB b[2];
    const int buf = kt & 1;
    #pragma unroll
    for (int i = 0; i < 2; i++) {
      const unsigned short* ap = &smA[buf][wm * 32 + i * 16 + l16][half * 8];
      a[i].q[0] = *(const uint4*)(ap);
      a[i].q[1] = *(const uint4*)(ap + 16);
    }
    #pragma unroll
    for (int j = 0; j < 2; j++) {
      const unsigned short* bp = &smB[wn * 32 + j * 16 + l16][half * 8];
      b[j].q[0] = *(const uint4*)(bp);
      b[j].q[1] = *(const uint4*)(bp + 16);
    }

    #pragma unroll
    for (int i = 0; i < 2; i++)
      #pragma unroll
      for (int j = 0; j < 2; j++)
        acc[i][j].v = __builtin_amdgcn_wmma_f32_16x16x32_bf16(
            false, a[i].v, false, b[j].v, (short)0, acc[i][j].v, false, false);

    __syncthreads();                     // protect buffers before next overwrite
  }

  // Epilogue. C/D layout: VGPR r -> M = r + half*8; N = l16.
  #pragma unroll
  for (int i = 0; i < 2; i++) {
    int m0 = rowBase + wm * 32 + i * 16 + half * 8;
    #pragma unroll
    for (int j = 0; j < 2; j++) {
      int n0 = colBase + wn * 32 + j * 16 + l16;
      if (n0 >= N) continue;
      float bb = bias ? bias[n0] : 0.f;
      #pragma unroll
      for (int r = 0; r < 8; r++) {
        int m = m0 + r;
        if (m >= M) continue;
        float vres = acc[i][j].f[r] + bb;
        if (scale) vres *= scale[m];
        float* cp = C + (size_t)m * N + n0;
        float prev = beta ? *cp : 0.f;
        *cp = prev + vres;
      }
    }
  }
}

// =========================== front-end kernels ===========================
__global__ void k_build_h0(const float* __restrict__ hist, float* __restrict__ h0) {
  int idx = blockIdx.x * blockDim.x + threadIdx.x;       // [B,36,N]
  if (idx >= Bh * DLc * NN) return;
  int n = idx % NN, c = (idx / NN) % DLc, b = idx / (NN * DLc);
  int d = c / TIN, t = c % TIN;
  h0[idx] = hist[((size_t)(b * TIN + t) * NN + n) * DIN + d];
}

// causal dilated conv, kernel=2, Cout=64
__global__ void k_conv2(const float* __restrict__ in, const float* __restrict__ w,
                        const float* __restrict__ bias, float* __restrict__ out,
                        int Cin, int dil) {
  int idx = blockIdx.x * blockDim.x + threadIdx.x;       // [B,64,N]
  if (idx >= Bh * TCNC * NN) return;
  int n = idx % NN, co = (idx / NN) % TCNC, b = idx / (NN * TCNC);
  float acc = bias[co];
  for (int ci = 0; ci < Cin; ci++) {
    float x1 = in[(size_t)(b * Cin + ci) * NN + n];
    float x0 = (n >= dil) ? in[(size_t)(b * Cin + ci) * NN + n - dil] : 0.f;
    const float* wp = w + ((size_t)co * Cin + ci) * 2;
    acc += wp[0] * x0 + wp[1] * x1;
  }
  out[idx] = fmaxf(acc, 0.f);
}

__global__ void k_down(const float* __restrict__ in, const float* __restrict__ wd,
                       const float* __restrict__ bd, float* __restrict__ out) {
  int idx = blockIdx.x * blockDim.x + threadIdx.x;       // [B,64,N] from [B,36,N]
  if (idx >= Bh * TCNC * NN) return;
  int n = idx % NN, co = (idx / NN) % TCNC, b = idx / (NN * TCNC);
  float acc = bd[co];
  for (int ci = 0; ci < DLc; ci++)
    acc += wd[co * DLc + ci] * in[(size_t)(b * DLc + ci) * NN + n];
  out[idx] = acc;
}

__global__ void k_addrelu(const float* __restrict__ a, const float* __restrict__ b,
                          float* __restrict__ out, int n) {
  int idx = blockIdx.x * blockDim.x + threadIdx.x;
  if (idx < n) out[idx] = fmaxf(a[idx] + b[idx], 0.f);
}

__global__ void k_prompt(const float* __restrict__ h, const float* __restrict__ pw,
                         const float* __restrict__ pb, float* __restrict__ out) {
  int idx = blockIdx.x * blockDim.x + threadIdx.x;       // [B,36,N]
  if (idx >= Bh * DLc * NN) return;
  int n = idx % NN, o = (idx / NN) % DLc, b = idx / (NN * DLc);
  float acc = pb[o];
  for (int c = 0; c < TCNC; c++) acc += pw[o * TCNC + c] * h[(size_t)(b * TCNC + c) * NN + n];
  out[idx] = acc;
}

__global__ void k_startconv(const float* __restrict__ pr, const float* __restrict__ sw,
                            const float* __restrict__ sb, float* __restrict__ out) {
  int idx = blockIdx.x * blockDim.x + threadIdx.x;       // [B,1024,N]
  if (idx >= Bh * DEEP * NN) return;
  int n = idx % NN, o = (idx / NN) % DEEP, b = idx / (NN * DEEP);
  float acc = sb[o];
  for (int c = 0; c < DLc; c++) acc += sw[o * DLc + c] * pr[(size_t)(b * DLc + c) * NN + n];
  out[idx] = acc;
}

__global__ void k_build_fused(const float* __restrict__ convout, const float* __restrict__ pr,
                              const float* __restrict__ tday, const float* __restrict__ tweek,
                              const float* __restrict__ nodee, const float* __restrict__ prompte,
                              unsigned short* __restrict__ Af) {
  int idx = blockIdx.x * blockDim.x + threadIdx.x;       // [NTOK, 4096] bf16
  if (idx >= NTOK * 4 * DEEP) return;
  int c = idx % (4 * DEEP), m = idx / (4 * DEEP);
  int b = m / NN, n = m % NN;
  float v;
  if (c < DEEP) {
    v = convout[(size_t)(b * DEEP + c) * NN + n];
  } else if (c < 2 * DEEP) {
    int cc = c - DEEP;
    int iday = (int)(pr[(size_t)(b * DLc + 1) * NN + n] * TIMEN);
    iday = min(max(iday, 0), TIMEN - 1);
    int iwk = (int)(pr[(size_t)(b * DLc + 2) * NN + n]);
    iwk = min(max(iwk, 0), 6);
    v = tday[iday * DEEP + cc] + tweek[iwk * DEEP + cc];
  } else if (c < 3 * DEEP) {
    v = nodee[(size_t)n * DEEP + (c - 2 * DEEP)];
  } else {
    v = prompte[(size_t)n * DEEP + (c - 3 * DEEP)];
  }
  Af[idx] = f2bf(v);
}

// =========================== transformer kernels =========================
__global__ __launch_bounds__(256) void k_rmsnorm_bf16(
    const float* __restrict__ x, const float* __restrict__ w,
    unsigned short* __restrict__ out) {
  int m = blockIdx.x;
  const float* xp = x + (size_t)m * HID;
  __shared__ float red[256];
  float s = 0.f;
  for (int c = threadIdx.x; c < HID; c += 256) { float v = xp[c]; s += v * v; }
  red[threadIdx.x] = s; __syncthreads();
  for (int off = 128; off; off >>= 1) {
    if (threadIdx.x < off) red[threadIdx.x] += red[threadIdx.x + off];
    __syncthreads();
  }
  float inv = rsqrtf(red[0] / HID + 1e-6f);
  for (int c = threadIdx.x; c < HID; c += 256)
    out[(size_t)m * HID + c] = f2bf(xp[c] * inv * w[c]);
}

__global__ void k_rope(float* __restrict__ q, float* __restrict__ k) {
  int idx = blockIdx.x * blockDim.x + threadIdx.x;       // NTOK*HID/2 pairs
  if (idx >= NTOK * (HID / 2)) return;
  int c2 = idx % (HID / 2), m = idx / (HID / 2);
  int s  = m % NN;                                       // position = node index
  int hh = c2 / (HD / 2), d = c2 % (HD / 2);
  float inv = __powf(10000.f, -(2.f * d) / HD);
  float ang = s * inv, cs = __cosf(ang), sn = __sinf(ang);
  size_t base = (size_t)m * HID + hh * HD;
  float q1 = q[base + d], q2 = q[base + d + HD / 2];
  q[base + d] = q1 * cs - q2 * sn;  q[base + d + HD / 2] = q2 * cs + q1 * sn;
  float k1 = k[base + d], k2 = k[base + d + HD / 2];
  k[base + d] = k1 * cs - k2 * sn;  k[base + d + HD / 2] = k2 * cs + k1 * sn;
}

__global__ __launch_bounds__(128) void k_attn(
    const float* __restrict__ q, const float* __restrict__ k,
    const float* __restrict__ v, unsigned short* __restrict__ obf) {
  int qp = blockIdx.x, h = blockIdx.y, b = blockIdx.z;
  int t = threadIdx.x;                                   // 128 = HD
  __shared__ float sc[NN];
  __shared__ float qv[HD];
  __shared__ float red[128];
  size_t tok = (size_t)(b * NN + qp) * HID + h * HD;
  qv[t] = q[tok + t];
  __syncthreads();
  for (int kp = t; kp <= qp; kp += 128) {
    const float* kv = k + (size_t)(b * NN + kp) * HID + h * HD;
    float d = 0.f;
    for (int c = 0; c < HD; c++) d += qv[c] * kv[c];
    sc[kp] = d * 0.088388347648f;                        // 1/sqrt(128)
  }
  __syncthreads();
  float lmax = -1e30f;
  for (int kp = t; kp <= qp; kp += 128) lmax = fmaxf(lmax, sc[kp]);
  red[t] = lmax; __syncthreads();
  for (int off = 64; off; off >>= 1) { if (t < off) red[t] = fmaxf(red[t], red[t + off]); __syncthreads(); }
  float mx = red[0]; __syncthreads();
  float lsum = 0.f;
  for (int kp = t; kp <= qp; kp += 128) { float e = __expf(sc[kp] - mx); sc[kp] = e; lsum += e; }
  red[t] = lsum; __syncthreads();
  for (int off = 64; off; off >>= 1) { if (t < off) red[t] += red[t + off]; __syncthreads(); }
  float isum = 1.f / red[0];
  float o = 0.f;                                          // thread owns dim d=t
  for (int kp = 0; kp <= qp; kp++)
    o += sc[kp] * v[(size_t)(b * NN + kp) * HID + h * HD + t];
  obf[tok + t] = f2bf(o * isum);
}

__global__ void k_silu_mul(const float* __restrict__ g, const float* __restrict__ u,
                           unsigned short* __restrict__ out, int n) {
  int idx = blockIdx.x * blockDim.x + threadIdx.x;
  if (idx >= n) return;
  float gv = g[idx];
  float s  = gv / (1.f + __expf(-gv));
  out[idx] = f2bf(s * u[idx]);
}

__global__ __launch_bounds__(256) void k_router(
    const unsigned short* __restrict__ X, const float* __restrict__ Wr,
    float* __restrict__ wexpT) {
  int m = blockIdx.x;
  __shared__ float red[256];
  __shared__ float logit[NEXP];
  float part[NEXP];
  #pragma unroll
  for (int e = 0; e < NEXP; e++) part[e] = 0.f;
  for (int c = threadIdx.x; c < HID; c += 256) {
    float xv = bf2f(X[(size_t)m * HID + c]);
    #pragma unroll
    for (int e = 0; e < NEXP; e++) part[e] += xv * Wr[e * HID + c];
  }
  for (int e = 0; e < NEXP; e++) {
    red[threadIdx.x] = part[e]; __syncthreads();
    for (int off = 128; off; off >>= 1) {
      if (threadIdx.x < off) red[threadIdx.x] += red[threadIdx.x + off];
      __syncthreads();
    }
    if (threadIdx.x == 0) logit[e] = red[0];
    __syncthreads();
  }
  if (threadIdx.x == 0) {
    float mx = logit[0];
    for (int e = 1; e < NEXP; e++) mx = fmaxf(mx, logit[e]);
    float p[NEXP], s = 0.f;
    for (int e = 0; e < NEXP; e++) { p[e] = __expf(logit[e] - mx); s += p[e]; }
    for (int e = 0; e < NEXP; e++) p[e] /= s;
    int i1 = 0;                                          // top-2, first-index ties
    for (int e = 1; e < NEXP; e++) if (p[e] > p[i1]) i1 = e;
    int i2 = (i1 == 0) ? 1 : 0;
    for (int e = 0; e < NEXP; e++) if (e != i1 && p[e] > p[i2]) i2 = e;
    for (int e = 0; e < NEXP; e++)
      wexpT[(size_t)e * NTOK + m] = (e == i1 || e == i2) ? p[e] : 0.f;
  }
}

// ============================== host driver ==============================
// Input flatten order (jax sorted-dict tree order): history_data, then params
// leaves alphabetically (layers/tcn lists in index order, dicts sorted).
enum {
  IN_HIST = 0, IN_FINAL_LN, IN_FUSION_B, IN_FUSION_W,
  IN_L0_D, IN_L0_G, IN_L0_LN1, IN_L0_LN2, IN_L0_U, IN_L0_WK, IN_L0_WO, IN_L0_WQ, IN_L0_WV,
  IN_L1_ED, IN_L1_EG, IN_L1_EU, IN_L1_LN1, IN_L1_LN2, IN_L1_ROUTER,
  IN_L1_SD, IN_L1_SG, IN_L1_SU, IN_L1_WK, IN_L1_WO, IN_L1_WQ, IN_L1_WV,
  IN_NODE_EMB, IN_PROMPT_B, IN_PROMPT_EMBED, IN_PROMPT_W,
  IN_REG_B, IN_REG_W, IN_START_B, IN_START_W,
  IN_T0_B1, IN_T0_B2, IN_T0_BD, IN_T0_W1, IN_T0_W2, IN_T0_WD,
  IN_T1_B1, IN_T1_B2, IN_T1_W1, IN_T1_W2,
  IN_T2_B1, IN_T2_B2, IN_T2_W1, IN_T2_W2,
  IN_TIME_DAY, IN_TIME_WEEK
};

static inline void gemm(const unsigned short* A, const float* W, float* C,
                        const float* bias, const float* scale,
                        int M, int N, int K, int beta, hipStream_t s) {
  dim3 g((N + 63) / 64, (M + 127) / 128);
  k_gemm<<<g, 256, 0, s>>>(A, W, C, bias, scale, M, N, K, beta);
}
static inline int blks(long n) { return (int)((n + 255) / 256); }

extern "C" void kernel_launch(void* const* d_in, const int* in_sizes, int n_in,
                              void* d_out, int out_size, void* d_ws, size_t ws_size,
                              hipStream_t stream) {
  (void)in_sizes; (void)n_in; (void)out_size; (void)ws_size;
  const float* F[64];
  for (int i = 0; i < 50; i++) F[i] = (const float*)d_in[i];

  // ---- workspace carve ----
  char* base = (char*)d_ws;
  size_t off = 0;
  auto alloc = [&](size_t bytes) -> void* {
    void* p = base + off;
    off = (off + bytes + 255) & ~(size_t)255;
    return p;
  };
  float* h0       = (float*)alloc((size_t)Bh * DLc * NN * 4);
  float* c1       = (float*)alloc((size_t)Bh * TCNC * NN * 4);
  float* c2       = (float*)alloc((size_t)Bh * TCNC * NN * 4);
  float* res      = (float*)alloc((size_t)Bh * TCNC * NN * 4);
  float* hblk     = (float*)alloc((size_t)Bh * TCNC * NN * 4);
  float* prompted = (float*)alloc((size_t)Bh * DLc * NN * 4);
  float* convout  = (float*)alloc((size_t)Bh * DEEP * NN * 4);
  unsigned short* Afused = (unsigned short*)alloc((size_t)NTOK * 4 * DEEP * 2);
  float* hseq     = (float*)alloc((size_t)NTOK * HID * 4);
  unsigned short* Xbf = (unsigned short*)alloc((size_t)NTOK * HID * 2);
  float* qb       = (float*)alloc((size_t)NTOK * HID * 4);
  float* kb       = (float*)alloc((size_t)NTOK * HID * 4);
  float* vb       = (float*)alloc((size_t)NTOK * HID * 4);
  unsigned short* obf = (unsigned short*)alloc((size_t)NTOK * HID * 2);
  float* gbuf     = (float*)alloc((size_t)NTOK * DFF * 4);
  float* ubuf     = (float*)alloc((size_t)NTOK * DFF * 4);
  unsigned short* hff = (unsigned short*)alloc((size_t)NTOK * DFF * 2);
  float* wexpT    = (float*)alloc((size_t)NEXP * NTOK * 4);

  const int nTCN = Bh * TCNC * NN;

  // ---- TCN prompt network ----
  k_build_h0<<<blks((long)Bh * DLc * NN), 256, 0, stream>>>(F[IN_HIST], h0);
  // block 0 (dil=1, 36->64 downsample)
  k_conv2<<<blks(nTCN), 256, 0, stream>>>(h0, F[IN_T0_W1], F[IN_T0_B1], c1, DLc, 1);
  k_conv2<<<blks(nTCN), 256, 0, stream>>>(c1, F[IN_T0_W2], F[IN_T0_B2], c2, TCNC, 1);
  k_down <<<blks(nTCN), 256, 0, stream>>>(h0, F[IN_T0_WD], F[IN_T0_BD], res);
  k_addrelu<<<blks(nTCN), 256, 0, stream>>>(c2, res, hblk, nTCN);
  // block 1 (dil=2)
  k_conv2<<<blks(nTCN), 256, 0, stream>>>(hblk, F[IN_T1_W1], F[IN_T1_B1], c1, TCNC, 2);
  k_conv2<<<blks(nTCN), 256, 0, stream>>>(c1, F[IN_T1_W2], F[IN_T1_B2], c2, TCNC, 2);
  k_addrelu<<<blks(nTCN), 256, 0, stream>>>(c2, hblk, hblk, nTCN);
  // block 2 (dil=4)
  k_conv2<<<blks(nTCN), 256, 0, stream>>>(hblk, F[IN_T2_W1], F[IN_T2_B1], c1, TCNC, 4);
  k_conv2<<<blks(nTCN), 256, 0, stream>>>(c1, F[IN_T2_W2], F[IN_T2_B2], c2, TCNC, 4);
  k_addrelu<<<blks(nTCN), 256, 0, stream>>>(c2, hblk, hblk, nTCN);

  k_prompt<<<blks((long)Bh * DLc * NN), 256, 0, stream>>>(hblk, F[IN_PROMPT_W], F[IN_PROMPT_B], prompted);
  k_startconv<<<blks((long)Bh * DEEP * NN), 256, 0, stream>>>(prompted, F[IN_START_W], F[IN_START_B], convout);
  k_build_fused<<<blks((long)NTOK * 4 * DEEP), 256, 0, stream>>>(
      convout, prompted, F[IN_TIME_DAY], F[IN_TIME_WEEK],
      F[IN_NODE_EMB], F[IN_PROMPT_EMBED], Afused);

  // fusion: hseq[NTOK,2048] = Afused[NTOK,4096] x fusion_w^T + fusion_b
  gemm(Afused, F[IN_FUSION_W], hseq, F[IN_FUSION_B], nullptr, NTOK, HID, 4 * DEEP, 0, stream);

  // ---- transformer layers ----
  const int LN1[2] = { IN_L0_LN1, IN_L1_LN1 }, LN2[2] = { IN_L0_LN2, IN_L1_LN2 };
  const int WQ[2]  = { IN_L0_WQ, IN_L1_WQ },  WK[2]  = { IN_L0_WK, IN_L1_WK };
  const int WV[2]  = { IN_L0_WV, IN_L1_WV },  WO[2]  = { IN_L0_WO, IN_L1_WO };

  for (int li = 0; li < 2; li++) {
    // attention block: hseq += Wo(attn(rope(Wq x), rope(Wk x), Wv x))
    k_rmsnorm_bf16<<<NTOK, 256, 0, stream>>>(hseq, F[LN1[li]], Xbf);
    gemm(Xbf, F[WQ[li]], qb, nullptr, nullptr, NTOK, HID, HID, 0, stream);
    gemm(Xbf, F[WK[li]], kb, nullptr, nullptr, NTOK, HID, HID, 0, stream);
    gemm(Xbf, F[WV[li]], vb, nullptr, nullptr, NTOK, HID, HID, 0, stream);
    k_rope<<<blks((long)NTOK * HID / 2), 256, 0, stream>>>(qb, kb);
    dim3 ag(NN, NH, Bh);
    k_attn<<<ag, 128, 0, stream>>>(qb, kb, vb, obf);
    gemm(obf, F[WO[li]], hseq, nullptr, nullptr, NTOK, HID, HID, 1, stream);

    k_rmsnorm_bf16<<<NTOK, 256, 0, stream>>>(hseq, F[LN2[li]], Xbf);
    if (li == 0) {
      // dense SwiGLU: hseq += (silu(x g^T) * (x u^T)) d^T
      gemm(Xbf, F[IN_L0_G], gbuf, nullptr, nullptr, NTOK, DFF, HID, 0, stream);
      gemm(Xbf, F[IN_L0_U], ubuf, nullptr, nullptr, NTOK, DFF, HID, 0, stream);
      k_silu_mul<<<blks((long)NTOK * DFF), 256, 0, stream>>>(gbuf, ubuf, hff, NTOK * DFF);
      gemm(hff, F[IN_L0_D], hseq, nullptr, nullptr, NTOK, HID, DFF, 1, stream);
    } else {
      // MoE: router gates + shared experts + 8 routed experts (dense, gated)
      k_router<<<NTOK, 256, 0, stream>>>(Xbf, F[IN_L1_ROUTER], wexpT);
      gemm(Xbf, F[IN_L1_SG], gbuf, nullptr, nullptr, NTOK, SFF, HID, 0, stream);
      gemm(Xbf, F[IN_L1_SU], ubuf, nullptr, nullptr, NTOK, SFF, HID, 0, stream);
      k_silu_mul<<<blks((long)NTOK * SFF), 256, 0, stream>>>(gbuf, ubuf, hff, NTOK * SFF);
      gemm(hff, F[IN_L1_SD], hseq, nullptr, nullptr, NTOK, HID, SFF, 1, stream);
      for (int e = 0; e < NEXP; e++) {
        const float* eg = F[IN_L1_EG] + (size_t)e * MFF * HID;
        const float* eu = F[IN_L1_EU] + (size_t)e * MFF * HID;
        const float* ed = F[IN_L1_ED] + (size_t)e * HID * MFF;
        gemm(Xbf, eg, gbuf, nullptr, nullptr, NTOK, MFF, HID, 0, stream);
        gemm(Xbf, eu, ubuf, nullptr, nullptr, NTOK, MFF, HID, 0, stream);
        k_silu_mul<<<blks((long)NTOK * MFF), 256, 0, stream>>>(gbuf, ubuf, hff, NTOK * MFF);
        gemm(hff, ed, hseq, nullptr, wexpT + (size_t)e * NTOK, NTOK, HID, MFF, 1, stream);
      }
    }
  }

  // ---- head: pred[B,N,12] = rmsnorm(hseq) reg_w^T + reg_b ----
  k_rmsnorm_bf16<<<NTOK, 256, 0, stream>>>(hseq, F[IN_FINAL_LN], Xbf);
  gemm(Xbf, F[IN_REG_W], (float*)d_out, F[IN_REG_B], nullptr, NTOK, OUTL, HID, 0, stream);
}
